// ImprovedSeqGNN_32349693673474
// MI455X (gfx1250) — matrix operand order repaired and checked by reading, compiled, not run on hardware
//
#include <hip/hip_runtime.h>
#include <hip/hip_bf16.h>

typedef __attribute__((ext_vector_type(16))) _Float16 v16h;
typedef __attribute__((ext_vector_type(8)))  float    v8f;
typedef __attribute__((ext_vector_type(4)))  int      v4i;

#define B_      128
#define T_      2048
#define G_      512      // 4*LH gates
#define K_      128      // LSTM hidden / K dim
#define KT_     4        // K tiles of 32
#define BT_     16       // batch rows per workgroup (= WMMA M)
#define THREADS 512      // 16 waves

// Branchless fast activations: keep the recurrence critical path on
// v_exp_f32 / v_rcp_f32 instead of libm's divergent tanhf expansion.
__device__ __forceinline__ float fast_sigmoid(float x) {
    return __builtin_amdgcn_rcpf(1.f + __expf(-x));
}
__device__ __forceinline__ float fast_tanh(float x) {
    float e = __expf(2.f * x);
    return 1.f - 2.f * __builtin_amdgcn_rcpf(e + 1.f);
}

// ---------------------------------------------------------------------------
// Fragment packing (CDNA5 wave32 WMMA f16 layouts, CDNA5 ISA)
//   A (16xK, 16-bit): lane = m + 16*((k&8)?1:0), elem = (k&7) + ((k&16)?8:0)   per 32-K tile
//   B (32x16, 16-bit): lane = n + 16*((k&8)?1:0), elem = (k&7) + ((k&16)?8:0)  per 32-K tile
// Packed global layout per B matrix [G gates x K]:
//   frag(nt,kt) at ((nt*KT + kt)*512 + lane*16 + elem) halves.
// ---------------------------------------------------------------------------

// Fold GCN + duplicated-input projection of LSTM0 into Wcomb[16,512] (packed as
// B fragments with K padded 16->32 with zeros) + bcomb.  Also bias1 = bih1+bhh1.
__global__ void prep_combined(const float* __restrict__ Wgcn,   // [16,64]
                              const float* __restrict__ bgcn,   // [64]
                              const float* __restrict__ Wih0,   // [512,128]
                              const float* __restrict__ bih0,
                              const float* __restrict__ bhh0,
                              const float* __restrict__ bih1,
                              const float* __restrict__ bhh1,
                              _Float16* __restrict__ wcombP,    // 32 nt * 512 halves
                              float* __restrict__ bcomb,        // [512]
                              float* __restrict__ bias1)        // [512]
{
    int j = blockIdx.x * blockDim.x + threadIdx.x;
    if (j >= G_) return;
    float weff[64];
    for (int e = 0; e < 64; ++e)
        weff[e] = Wih0[j * 128 + e] + Wih0[j * 128 + 64 + e];
    float bc = bih0[j] + bhh0[j];
    for (int e = 0; e < 64; ++e) bc += bgcn[e] * weff[e];
    bcomb[j] = bc;
    bias1[j] = bih1[j] + bhh1[j];

    int nt = j >> 4, n = j & 15;
    for (int d = 0; d < 16; ++d) {
        float v = 0.f;
        for (int e = 0; e < 64; ++e) v += Wgcn[d * 64 + e] * weff[e];
        wcombP[nt * 512 + (n + ((d & 8) ? 16 : 0)) * 16 + (d & 7)] = (_Float16)v;          // k=d<16: lane=n+16*(d>=8), elem=d&7
        wcombP[nt * 512 + (n + ((d & 8) ? 16 : 0)) * 16 + (d & 7) + 8] = (_Float16)0.f; // k=d+16 zero pad
    }
}

// Pack a [512,128] f32 weight (rows = gate/N, cols = K) into f16 B-fragments.
__global__ void pack_b512x128(const float* __restrict__ src, _Float16* __restrict__ dst)
{
    int idx = blockIdx.x * blockDim.x + threadIdx.x;   // 65536 elements
    if (idx >= G_ * K_) return;
    int j  = idx >> 7, k = idx & 127;
    int nt = j >> 4, n = j & 15;
    int kt = k >> 5, kk = k & 31;
    int lane = n + ((kk & 8) ? 16 : 0);
    int e    = (kk & 7) + ((kk & 16) ? 8 : 0);
    dst[(nt * KT_ + kt) * 512 + lane * 16 + e] = (_Float16)src[j * 128 + k];
}

// ---------------------------------------------------------------------------
// Persistent fused scan kernel. Dynamic LDS map (bytes):
//   [0      )  h0 A-fragments   : 2048 halves (4 K-tiles * 512)
//   [4096   )  h1 A-fragments   : 2048 halves
//   [8192   )  xs A-fragment    : 512 halves (K padded to 32)
//   [9216   )  gates0 f32       : 16*512
//   [41984  )  gates1 f32       : 16*512
//   [74752  )  h1 final f32     : 16*128
//   [82944  )  Wih1 B-fragments : 65536 halves  (131072 B)  -> total 214016 B
// ---------------------------------------------------------------------------
__launch_bounds__(THREADS, 1)
__global__ void seq_gnn_lstm(const float* __restrict__ x,
                             const _Float16* __restrict__ wcombP,
                             const _Float16* __restrict__ whh0P,
                             const _Float16* __restrict__ wih1P,
                             const _Float16* __restrict__ whh1P,
                             const float* __restrict__ bcomb,
                             const float* __restrict__ bias1,
                             const float* __restrict__ W1, const float* __restrict__ b1,
                             const float* __restrict__ W2, const float* __restrict__ b2,
                             const float* __restrict__ W3, const float* __restrict__ b3,
                             float* __restrict__ out)
{
    extern __shared__ char smem[];
    _Float16* h0f   = (_Float16*)(smem);
    _Float16* h1f   = (_Float16*)(smem + 4096);
    _Float16* xsf   = (_Float16*)(smem + 8192);
    float*    g0    = (float*)(smem + 9216);
    float*    g1    = (float*)(smem + 41984);
    float*    hfin  = (float*)(smem + 74752);
    _Float16* wih1L = (_Float16*)(smem + 82944);

    const int tid  = threadIdx.x;
    const int wave = tid >> 5;
    const int lane = tid & 31;
    const int b0   = blockIdx.x * BT_;

    // zero hidden-state fragments (h0f/h1f contiguous: 4096 halves)
    for (int i = tid; i < 4096; i += THREADS) h0f[i] = (_Float16)0.f;

    // ---- stage Wih1 fragments into LDS (128 KB) ----
#if __has_builtin(__builtin_amdgcn_global_load_async_to_lds_b128) && \
    __has_builtin(__builtin_amdgcn_s_wait_asynccnt)
    {
        // gfx1250 async copy: 16 B/lane per issue, ASYNCcnt-tracked, no VGPR bounce.
        char*       srcb = (char*)wih1P;   // const cast away: builtin takes v4i*
        char*       dstb = (char*)wih1L;
        for (int i = tid; i < (G_ * K_ * 2) / 16; i += THREADS) {
            __builtin_amdgcn_global_load_async_to_lds_b128(
                (v4i*)(srcb + (size_t)i * 16), (v4i*)(dstb + (size_t)i * 16), 0, 0);
        }
        __builtin_amdgcn_s_wait_asynccnt(0);
    }
#else
    {
        const uint4* src = (const uint4*)wih1P;
        uint4*       dst = (uint4*)wih1L;
        for (int i = tid; i < (G_ * K_ * 2) / 16; i += THREADS) dst[i] = src[i];
    }
#endif

    // per-wave weight fragments held in VGPRs
    v16h wcF[2], whh0F[2][KT_], whh1F[2][KT_];
    float bb0[2], bb1[2];
    for (int nti = 0; nti < 2; ++nti) {
        int nt = wave * 2 + nti;
        wcF[nti] = *(const v16h*)(wcombP + nt * 512 + lane * 16);
        for (int kt = 0; kt < KT_; ++kt) {
            whh0F[nti][kt] = *(const v16h*)(whh0P + (nt * KT_ + kt) * 512 + lane * 16);
            whh1F[nti][kt] = *(const v16h*)(whh1P + (nt * KT_ + kt) * 512 + lane * 16);
        }
        bb0[nti] = bcomb[nt * 16 + (lane & 15)];
        bb1[nti] = bias1[nt * 16 + (lane & 15)];
    }

    // activation ownership: thread -> row rAct, columns j0..j0+3 (cell state in regs)
    const int rAct = tid >> 5;
    const int j0   = (tid & 31) * 4;
    float c0[4] = {0.f, 0.f, 0.f, 0.f};
    float c1[4] = {0.f, 0.f, 0.f, 0.f};

    // xs producer: xs = 0.5*(x[...,:16] + x[...,16:]) packed as A fragment, K 16..31 zero
    auto produce_xs = [&](int t) {
        if (tid < 256) {
            int r = tid >> 4, k = tid & 15;
            const float* xr = x + ((size_t)(b0 + r) * T_ + t) * 32;
            float v = 0.5f * (xr[k] + xr[k + 16]);
            int laneA = r + ((k & 8) ? 16 : 0);
            int e     = (k & 7);
            xsf[laneA * 16 + e]     = (_Float16)v;
            xsf[laneA * 16 + e + 8] = (_Float16)0.f;   // zero pad for k+16
            if (t + 1 < T_) __builtin_prefetch(xr + 32, 0, 1);
        }
    };

    produce_xs(0);
    __syncthreads();

    for (int t = 0; t < T_; ++t) {
        // -------- layer 0 gates: xs@Wcomb + h0@Whh0^T + bcomb --------
        v16h xsA = *(const v16h*)(xsf + lane * 16);
        v16h h0A[KT_];
        for (int kt = 0; kt < KT_; ++kt)
            h0A[kt] = *(const v16h*)(h0f + kt * 512 + lane * 16);
        for (int nti = 0; nti < 2; ++nti) {
            v8f acc;
            for (int i = 0; i < 8; ++i) acc[i] = bb0[nti];
            acc = __builtin_amdgcn_wmma_f32_16x16x32_f16(false, xsA, false, wcF[nti],
                                                         (short)0, acc, false, false);
            for (int kt = 0; kt < KT_; ++kt)
                acc = __builtin_amdgcn_wmma_f32_16x16x32_f16(false, h0A[kt], false,
                                                             whh0F[nti][kt],
                                                             (short)0, acc, false, false);
            int m0 = (lane >= 16) ? 8 : 0;
            int col = (wave * 2 + nti) * 16 + (lane & 15);
            for (int i = 0; i < 8; ++i) g0[(m0 + i) * G_ + col] = acc[i];
        }
        __syncthreads();

        // -------- layer 0 activations -> h0 fragment --------
        for (int u = 0; u < 4; ++u) {
            int j = j0 + u;
            float gi = g0[rAct * G_ + j];
            float gf = g0[rAct * G_ + 128 + j];
            float gg = g0[rAct * G_ + 256 + j];
            float go = g0[rAct * G_ + 384 + j];
            float c  = fast_sigmoid(gf) * c0[u] + fast_sigmoid(gi) * fast_tanh(gg);
            c0[u] = c;
            float h = fast_sigmoid(go) * fast_tanh(c);
            int kt = j >> 5, kk = j & 31;
            int laneA = rAct + ((kk & 8) ? 16 : 0);
            int e     = (kk & 7) + ((kk & 16) ? 8 : 0);
            h0f[kt * 512 + laneA * 16 + e] = (_Float16)h;
        }
        __syncthreads();

        // -------- layer 1 gates: h0_new@Wih1^T + h1@Whh1^T + bias1 --------
        v16h h0B[KT_], h1A[KT_];
        for (int kt = 0; kt < KT_; ++kt) {
            h0B[kt] = *(const v16h*)(h0f + kt * 512 + lane * 16);
            h1A[kt] = *(const v16h*)(h1f + kt * 512 + lane * 16);
        }
        for (int nti = 0; nti < 2; ++nti) {
            int nt = wave * 2 + nti;
            v8f acc;
            for (int i = 0; i < 8; ++i) acc[i] = bb1[nti];
            for (int kt = 0; kt < KT_; ++kt) {
                v16h wb = *(const v16h*)(wih1L + (nt * KT_ + kt) * 512 + lane * 16);
                acc = __builtin_amdgcn_wmma_f32_16x16x32_f16(false, h0B[kt], false, wb,
                                                             (short)0, acc, false, false);
                acc = __builtin_amdgcn_wmma_f32_16x16x32_f16(false, h1A[kt], false,
                                                             whh1F[nti][kt],
                                                             (short)0, acc, false, false);
            }
            int m0 = (lane >= 16) ? 8 : 0;
            int col = nt * 16 + (lane & 15);
            for (int i = 0; i < 8; ++i) g1[(m0 + i) * G_ + col] = acc[i];
        }
        __syncthreads();

        // -------- layer 1 activations -> h1 fragment; overlap xs(t+1) --------
        for (int u = 0; u < 4; ++u) {
            int j = j0 + u;
            float gi = g1[rAct * G_ + j];
            float gf = g1[rAct * G_ + 128 + j];
            float gg = g1[rAct * G_ + 256 + j];
            float go = g1[rAct * G_ + 384 + j];
            float c  = fast_sigmoid(gf) * c1[u] + fast_sigmoid(gi) * fast_tanh(gg);
            c1[u] = c;
            float h = fast_sigmoid(go) * fast_tanh(c);
            int kt = j >> 5, kk = j & 31;
            int laneA = rAct + ((kk & 8) ? 16 : 0);
            int e     = (kk & 7) + ((kk & 16) ? 8 : 0);
            h1f[kt * 512 + laneA * 16 + e] = (_Float16)h;
            if (t == T_ - 1) hfin[rAct * 128 + j] = h;
        }
        if (t + 1 < T_) produce_xs(t + 1);
        __syncthreads();
    }

    // -------- fused MLP head (reuse gate buffers as scratch) --------
    float* s1 = g0;   // [16,128]
    for (int idx = tid; idx < 16 * 128; idx += THREADS) {
        int r = idx >> 7, o = idx & 127;
        float a = b1[o];
        const float* wr = W1 + o * 128;
        for (int k = 0; k < 128; ++k) a += hfin[r * 128 + k] * wr[k];
        s1[idx] = a > 0.f ? a : 0.f;
    }
    __syncthreads();
    float* s2 = g1;   // [16,64]
    for (int idx = tid; idx < 16 * 64; idx += THREADS) {
        int r = idx >> 6, o = idx & 63;
        float a = b2[o];
        const float* wr = W2 + o * 128;
        for (int k = 0; k < 128; ++k) a += s1[r * 128 + k] * wr[k];
        s2[idx] = a > 0.f ? a : 0.f;
    }
    __syncthreads();
    for (int idx = tid; idx < 16 * 26; idx += THREADS) {
        int r = idx / 26, o = idx % 26;
        float a = b3[o];
        const float* wr = W3 + o * 64;
        for (int k = 0; k < 64; ++k) a += s2[r * 64 + k] * wr[k];
        out[(b0 + r) * 26 + o] = a;
    }
}

// ---------------------------------------------------------------------------
extern "C" void kernel_launch(void* const* d_in, const int* in_sizes, int n_in,
                              void* d_out, int out_size, void* d_ws, size_t ws_size,
                              hipStream_t stream) {
    (void)in_sizes; (void)n_in; (void)out_size; (void)ws_size;
    const float* x    = (const float*)d_in[0];
    const float* Wgcn = (const float*)d_in[1];
    const float* bgcn = (const float*)d_in[2];
    const float* Wih0 = (const float*)d_in[3];
    const float* Whh0 = (const float*)d_in[4];
    const float* bih0 = (const float*)d_in[5];
    const float* bhh0 = (const float*)d_in[6];
    const float* Wih1 = (const float*)d_in[7];
    const float* Whh1 = (const float*)d_in[8];
    const float* bih1 = (const float*)d_in[9];
    const float* bhh1 = (const float*)d_in[10];
    const float* W1   = (const float*)d_in[11];
    const float* b1   = (const float*)d_in[12];
    const float* W2   = (const float*)d_in[13];
    const float* b2   = (const float*)d_in[14];
    const float* W3   = (const float*)d_in[15];
    const float* b3   = (const float*)d_in[16];

    char* ws = (char*)d_ws;
    _Float16* wcombP = (_Float16*)(ws + 0);        //  32 KB
    _Float16* whh0P  = (_Float16*)(ws + 32768);    // 128 KB
    _Float16* wih1P  = (_Float16*)(ws + 163840);   // 128 KB
    _Float16* whh1P  = (_Float16*)(ws + 294912);   // 128 KB
    float*    bcomb  = (float*)(ws + 425984);      //   2 KB
    float*    bias1  = (float*)(ws + 428032);      //   2 KB

    prep_combined<<<1, 512, 0, stream>>>(Wgcn, bgcn, Wih0, bih0, bhh0, bih1, bhh1,
                                         wcombP, bcomb, bias1);
    pack_b512x128<<<256, 256, 0, stream>>>(Whh0, whh0P);
    pack_b512x128<<<256, 256, 0, stream>>>(Wih1, wih1P);
    pack_b512x128<<<256, 256, 0, stream>>>(Whh1, whh1P);

    const size_t ldsBytes = 214016;   // < 320 KB WGP limit on CDNA5
    (void)hipFuncSetAttribute((const void*)seq_gnn_lstm,
                              hipFuncAttributeMaxDynamicSharedMemorySize, (int)ldsBytes);
    seq_gnn_lstm<<<B_ / BT_, THREADS, ldsBytes, stream>>>(
        x, wcombP, whh0P, wih1P, whh1P, bcomb, bias1,
        W1, b1, W2, b2, W3, b3, (float*)d_out);
}
